// QFFRadianceField_12223476924832
// MI455X (gfx1250) — matrix operand order repaired and checked by reading, not compiled, not run on hardware
//
#include <hip/hip_runtime.h>
#include <cstdint>
#include <cstddef>

typedef __attribute__((ext_vector_type(16))) _Float16 v16h;
typedef __attribute__((ext_vector_type(8)))  _Float16 v8h;
typedef __attribute__((ext_vector_type(4)))  _Float16 v4h;
typedef __attribute__((ext_vector_type(8)))  float    v8f;

#define FRAG 512  // halves per 16x16(x32) fragment: 32 lanes * 16 halves

// fragment offsets inside sW (units: halves)
#define OFF_G1 0             // 96x64 : 3 kc * 4 t = 12 frags
#define OFF_G2 (12*FRAG)     // 64x16 : 2 kc * 1 t
#define OFF_C1 (14*FRAG)     // 32x64 : 1 kc * 4 t (padded from 31 rows)
#define OFF_C2 (18*FRAG)     // 64x64 : 2 kc * 4 t
#define OFF_C3 (26*FRAG)     // 64x16 : 2 kc * 1 t (padded from 3 cols)
#define W_TOTAL (28*FRAG)

__device__ __forceinline__ v8f wmma_f16(v16h a, v16h b, v8f c) {
  // D = A(16x32,f16) x B(32x16,f16) + C(16x16,f32)
  return __builtin_amdgcn_wmma_f32_16x16x32_f16(false, a, false, b, (short)0, c,
                                                false, false);
}

// Store weight element W[k][n] into LDS in B-fragment order:
// lane = ((k%32 >> 3) & 1)*16 + (n%16), element = (k%8) + (k%32 >= 16 ? 8 : 0).
__device__ __forceinline__ void stashB(_Float16* base, int numT, int k, int n,
                                       float val) {
  int kc = k >> 5, kl = k & 31;
  int t = n >> 4;
  int lane = (((kl >> 3) & 1) << 4) + (n & 15);
  int e = (kl & 7) + ((kl & 16) ? 8 : 0);
  base[(kc * numT + t) * FRAG + lane * 16 + e] = (_Float16)val;
}

// Scatter a D accumulator tile (16 samples x 16 neurons, this wave) into the
// activation staging area laid out as next layer's A fragments (f16).
// A layout (16-bit 16x32): lane = g*16 + m, where for k%32 = kl:
//   g = (kl>>3)&1, v = ((kl&7)>>1) + (kl&16 ? 4 : 0), element = 2v + (kl&1).
__device__ __forceinline__ void storeDtoA(_Float16* act, v8f d, int t, int lane,
                                          bool relu) {
  int k = t * 16 + (lane & 15);      // this lane's output-neuron index = next-K
  int kc = k >> 5, kl = k & 31;
  int g = (kl >> 3) & 1;
  int v = ((kl & 7) >> 1) + ((kl & 16) ? 4 : 0);
  int e = 2 * v + (kl & 1);
  int mb = (lane >> 4) * 8;          // this lane holds samples mb..mb+7
  _Float16* p = act + kc * FRAG + e;
#pragma unroll
  for (int r = 0; r < 8; ++r) {
    float x = d[r];
    if (relu) x = fmaxf(x, 0.0f);
    p[((g << 4) + mb + r) * 16] = (_Float16)x;
  }
}

__global__ __launch_bounds__(256) void qff_radiance_kernel(
    const float* __restrict__ positions,   // (N,3)
    const float* __restrict__ directions,  // (N,3)
    const float* __restrict__ grid,        // (24,80,4)
    const float* __restrict__ w_geom1,     // (96,64)
    const float* __restrict__ w_geom2,     // (64,16)
    const float* __restrict__ w_color1,    // (31,64)
    const float* __restrict__ w_color2,    // (64,64)
    const float* __restrict__ w_color3,    // (64,3)
    float* __restrict__ out,               // rgb (N,3) then density (N,)
    int N) {
  __shared__ __align__(64) _Float16 sW[W_TOTAL];        // 28 KB
  __shared__ __align__(64) _Float16 sAct[8 * 3 * FRAG]; // 24 KB (3 KB / wave)

  int tid = threadIdx.x;

  // ---- cooperative weight conversion f32 -> f16, pre-swizzled B fragments ----
  for (int i = tid; i < 96 * 64; i += 256)
    stashB(sW + OFF_G1, 4, i >> 6, i & 63, w_geom1[i]);
  for (int i = tid; i < 64 * 16; i += 256)
    stashB(sW + OFF_G2, 1, i >> 4, i & 15, w_geom2[i]);
  for (int i = tid; i < 32 * 64; i += 256) {
    int k = i >> 6, n = i & 63;
    stashB(sW + OFF_C1, 4, k, n, (k < 31) ? w_color1[k * 64 + n] : 0.0f);
  }
  for (int i = tid; i < 64 * 64; i += 256)
    stashB(sW + OFF_C2, 4, i >> 6, i & 63, w_color2[i]);
  for (int i = tid; i < 64 * 16; i += 256) {
    int k = i >> 4, n = i & 15;
    stashB(sW + OFF_C3, 1, k, n, (n < 3) ? w_color3[k * 3 + n] : 0.0f);
  }
  __syncthreads();

  int wv = tid >> 5, lane = tid & 31;
  int m = lane & 15;   // sample within tile this lane reads/encodes
  int hh = lane >> 4;  // half-wave id
  int tile = blockIdx.x * 8 + wv;
  if (tile * 16 >= N) return;  // wave-uniform guard (EXEC stays all-ones)

  int sBase = tile * 16;
  int s = sBase + m;
  _Float16* act = sAct + wv * 3 * FRAG;

  // ================= QFF positional encoding (16 x 96 latent) ==============
  float px = positions[3 * s + 0];
  float py = positions[3 * s + 1];
  float pz = positions[3 * s + 2];
#pragma unroll
  for (int j = 0; j < 12; ++j) {
    int l = hh * 12 + j;           // line index 0..23
    int c = l / 3, dim = l - 3 * c;
    float pv = (dim == 0) ? px : ((dim == 1) ? py : pz);
    int cf = c & 3;                // freq = 2^linspace(1,6,4)
    float fr = (cf == 0) ? 2.0f
             : (cf == 1) ? 6.3496042078728f
             : (cf == 2) ? 20.1587367983318f
                         : 64.0f;
    float val = pv * fr;
    float coord = (c < 4) ? __sinf(val) : __cosf(val);
    float u = (coord + 1.0f) * 39.5f;  // * 0.5 * (80-1)
    float fi = fminf(fmaxf(floorf(u), 0.0f), 78.0f);
    float w = u - fi;
    const float* gp = grid + ((l * 80 + (int)fi) << 2);
    float4 g0 = *(const float4*)gp;
    float4 g1 = *(const float4*)(gp + 4);
    v4h feat;
    feat.x = (_Float16)(g0.x + (g1.x - g0.x) * w);
    feat.y = (_Float16)(g0.y + (g1.y - g0.y) * w);
    feat.z = (_Float16)(g0.z + (g1.z - g0.z) * w);
    feat.w = (_Float16)(g0.w + (g1.w - g0.w) * w);
    // write 4 latents (k = 4l..4l+3) at their A-fragment position
    int k0 = l * 4, kc = k0 >> 5, kl = k0 & 31;
    int g = (kl >> 3) & 1;
    int v0 = ((kl & 7) >> 1) + ((kl & 16) ? 4 : 0);
    *(v4h*)(act + kc * FRAG + (((g << 4) + m) * 16 + 2 * v0)) = feat;
  }
  asm volatile("s_wait_dscnt 0x0" ::: "memory");

  // ================= geom1: 16x96 @ 96x64, relu (12 WMMA) ==================
  v16h a0 = *(const v16h*)(act + lane * 16);
  v16h a1 = *(const v16h*)(act + FRAG + lane * 16);
  v16h a2 = *(const v16h*)(act + 2 * FRAG + lane * 16);
  {
    v8f acc[4] = {};
#pragma unroll
    for (int t = 0; t < 4; ++t) {
      acc[t] = wmma_f16(a0, *(const v16h*)(sW + OFF_G1 + (0 * 4 + t) * FRAG + lane * 16), acc[t]);
      acc[t] = wmma_f16(a1, *(const v16h*)(sW + OFF_G1 + (1 * 4 + t) * FRAG + lane * 16), acc[t]);
      acc[t] = wmma_f16(a2, *(const v16h*)(sW + OFF_G1 + (2 * 4 + t) * FRAG + lane * 16), acc[t]);
    }
#pragma unroll
    for (int t = 0; t < 4; ++t) storeDtoA(act, acc[t], t, lane, true);
  }
  asm volatile("s_wait_dscnt 0x0" ::: "memory");

  // ================= geom2: 16x64 @ 64x16 (2 WMMA) =========================
  a0 = *(const v16h*)(act + lane * 16);
  a1 = *(const v16h*)(act + FRAG + lane * 16);
  v8f f = {};
  f = wmma_f16(a0, *(const v16h*)(sW + OFF_G2 + 0 * FRAG + lane * 16), f);
  f = wmma_f16(a1, *(const v16h*)(sW + OFF_G2 + 1 * FRAG + lane * 16), f);

  // ---- build color input hc = [SH16 | feature15 | 0] as A fragment (k<32) --
  float dx = directions[3 * s + 0];
  float dy = directions[3 * s + 1];
  float dz = directions[3 * s + 2];
  float xx = dx * dx, yy = dy * dy, zz = dz * dz;
  float xy = dx * dy, yz = dy * dz, xz = dx * dz;
  v8h shv;
  if (hh == 0) {  // SH 0..7 -> k 0..7 (g=0, e=k), this lane's own slot
    shv[0] = (_Float16)0.28209479177387814f;
    shv[1] = (_Float16)(-0.48860251190291987f * dy);
    shv[2] = (_Float16)( 0.48860251190291987f * dz);
    shv[3] = (_Float16)(-0.48860251190291987f * dx);
    shv[4] = (_Float16)( 1.0925484305920792f * xy);
    shv[5] = (_Float16)(-1.0925484305920792f * yz);
    shv[6] = (_Float16)( 0.94617469575756f * zz - 0.31539156525252f);
    shv[7] = (_Float16)(-1.0925484305920792f * xz);
  } else {        // SH 8..15 -> k 8..15 (g=1, e=k-8), own slot
    shv[0] = (_Float16)( 0.5462742152960396f * (xx - yy));
    shv[1] = (_Float16)(-0.5900435899266435f * dy * (3.0f * xx - yy));
    shv[2] = (_Float16)( 2.890611442640554f * xy * dz);
    shv[3] = (_Float16)(-0.4570457994644657f * dy * (4.0f * zz - xx - yy));
    shv[4] = (_Float16)( 0.37317633259011546f * dz * (2.0f * zz - 3.0f * xx - 3.0f * yy));
    shv[5] = (_Float16)(-0.4570457994644657f * dx * (4.0f * zz - xx - yy));
    shv[6] = (_Float16)( 1.445305721320277f * dz * (xx - yy));
    shv[7] = (_Float16)(-0.5900435899266435f * dx * (xx - 3.0f * yy));
  }
  *(v8h*)(act + lane * 16) = shv;  // elements 0..7 of this lane's slot

  int col = lane & 15;  // neuron column this lane holds in D tiles
  int mb = hh * 8;      // samples mb..mb+7 in this lane's accumulator
  if (col == 0) {
    // density = trunc_exp(f[:,0] - 1); also zero-fill k=31 pad slot
#pragma unroll
    for (int r = 0; r < 8; ++r) {
      out[(size_t)3 * N + sBase + mb + r] = __expf(f[r] - 1.0f);
      act[(16 + mb + r) * 16 + 15] = (_Float16)0.0f;  // k=31 (g=1, e=15)
    }
  } else {
    // feature j = col-1 -> k = 15 + col (16..30)
    int k = 15 + col;
    int g = (k >> 3) & 1;
    int v = ((k & 7) >> 1) + ((k & 16) ? 4 : 0);
    int e = 2 * v + (k & 1);
#pragma unroll
    for (int r = 0; r < 8; ++r)
      act[((g << 4) + mb + r) * 16 + e] = (_Float16)f[r];
  }
  asm volatile("s_wait_dscnt 0x0" ::: "memory");

  // ================= color1: 16x32 @ 32x64, relu (4 WMMA) ==================
  {
    v16h ca = *(const v16h*)(act + lane * 16);
    v8f c1[4] = {};
#pragma unroll
    for (int t = 0; t < 4; ++t)
      c1[t] = wmma_f16(ca, *(const v16h*)(sW + OFF_C1 + t * FRAG + lane * 16), c1[t]);
#pragma unroll
    for (int t = 0; t < 4; ++t) storeDtoA(act, c1[t], t, lane, true);
  }
  asm volatile("s_wait_dscnt 0x0" ::: "memory");

  // ================= color2: 16x64 @ 64x64, relu (8 WMMA) ==================
  a0 = *(const v16h*)(act + lane * 16);
  a1 = *(const v16h*)(act + FRAG + lane * 16);
  {
    v8f c2[4] = {};
#pragma unroll
    for (int t = 0; t < 4; ++t) {
      c2[t] = wmma_f16(a0, *(const v16h*)(sW + OFF_C2 + (0 * 4 + t) * FRAG + lane * 16), c2[t]);
      c2[t] = wmma_f16(a1, *(const v16h*)(sW + OFF_C2 + (1 * 4 + t) * FRAG + lane * 16), c2[t]);
    }
#pragma unroll
    for (int t = 0; t < 4; ++t) storeDtoA(act, c2[t], t, lane, true);
  }
  asm volatile("s_wait_dscnt 0x0" ::: "memory");

  // ================= color3: 16x64 @ 64x16 (2 WMMA), sigmoid ===============
  a0 = *(const v16h*)(act + lane * 16);
  a1 = *(const v16h*)(act + FRAG + lane * 16);
  v8f o = {};
  o = wmma_f16(a0, *(const v16h*)(sW + OFF_C3 + 0 * FRAG + lane * 16), o);
  o = wmma_f16(a1, *(const v16h*)(sW + OFF_C3 + 1 * FRAG + lane * 16), o);
  if (col < 3) {
#pragma unroll
    for (int r = 0; r < 8; ++r) {
      float x = o[r];
      out[(size_t)(sBase + mb + r) * 3 + col] = 1.0f / (1.0f + __expf(-x));
    }
  }
}

extern "C" void kernel_launch(void* const* d_in, const int* in_sizes, int n_in,
                              void* d_out, int out_size, void* d_ws,
                              size_t ws_size, hipStream_t stream) {
  const float* positions  = (const float*)d_in[0];
  const float* directions = (const float*)d_in[1];
  const float* grid       = (const float*)d_in[2];
  const float* w_geom1    = (const float*)d_in[3];
  const float* w_geom2    = (const float*)d_in[4];
  const float* w_color1   = (const float*)d_in[5];
  const float* w_color2   = (const float*)d_in[6];
  const float* w_color3   = (const float*)d_in[7];
  float* out = (float*)d_out;

  int N = in_sizes[0] / 3;           // positions is (N,3)
  int tiles = (N + 15) / 16;         // 16 samples per wave
  int blocks = (tiles + 7) / 8;      // 8 waves per 256-thread block
  qff_radiance_kernel<<<blocks, 256, 0, stream>>>(
      positions, directions, grid, w_geom1, w_geom2, w_color1, w_color2,
      w_color3, out, N);
}
